// FusionModel_22230750724550
// MI455X (gfx1250) — compile-verified, matrix operand order checked by reading, not run on hardware
//
#include <hip/hip_runtime.h>
#include <hip/hip_fp16.h>
#include <cstdint>

typedef __attribute__((ext_vector_type(16))) _Float16 v16h;
typedef __attribute__((ext_vector_type(8)))  _Float16 v8h;
typedef __attribute__((ext_vector_type(8)))  float    v8f;
typedef unsigned int u32x4 __attribute__((ext_vector_type(4)));
typedef int          i32x4 __attribute__((ext_vector_type(4)));
typedef int          i32x8 __attribute__((ext_vector_type(8)));

#define EMB    256
#define D0R    66      // real K of encoder input
#define K1PAD  96      // 66 -> 96 (3 K-tiles of 32)
#define KMPAD  288     // 258 -> 288 (9 K-tiles)
#define KUPAD  512     // 16 K-tiles
#define NDEC   1056    // MAX_OBJ * D0 = 16*66, 66 N-tiles
#define NTDEC  66

// ---- gfx1250 async / TDM feature detection (device pass only) --------------
#if defined(__HIP_DEVICE_COMPILE__)
#if __has_builtin(__builtin_amdgcn_global_load_async_to_lds_b128)
#define HAS_ASYNC_COPY 1
#endif
#if __has_builtin(__builtin_amdgcn_tensor_load_to_lds)
#define HAS_TDM 1
#endif
#endif

// builtin signature (from hipcc diagnostic): (v4i AS1*, v4i AS3*, imm, imm)
#define GPTR(p) ((__attribute__((address_space(1))) i32x4*)(p))
#define LPTR(p) ((__attribute__((address_space(3))) i32x4*)(p))

// ---------------------------------------------------------------------------
// Swizzled f16 B-tile layout (shared by converter and GEMM consumers).
// Tile t covers k in [kt*32, kt*32+32), n in [nt*16, nt*16+16).
// 512 halves per tile: lane l (0..31), elem i (0..15):
//   half = l>>4 ; n = nt*16 + (l&15)
//   k_local = (i<8) ? half*8 + i : 16 + half*8 + (i-8)
// => per lane, B fragment is 16 contiguous halves at tileBase + l*16.
// ---------------------------------------------------------------------------

__global__ void k_zero_f32(float* __restrict__ p, long long n) {
  long long i = (long long)blockIdx.x * blockDim.x + threadIdx.x;
  if (i < n) p[i] = 0.0f;
}

__global__ void k_f32_to_f16(const float* __restrict__ in,
                             _Float16* __restrict__ out, long long n) {
  long long i = (long long)blockIdx.x * blockDim.x + threadIdx.x;
  if (i < n) out[i] = (_Float16)in[i];
}

__global__ void k_convert(const float* __restrict__ W, _Float16* __restrict__ out,
                          int Kreal, int N, long long total) {
  long long idx = (long long)blockIdx.x * blockDim.x + threadIdx.x;
  if (idx >= total) return;
  const int ntiles = N >> 4;
  int t = (int)(idx >> 9);
  int w = (int)(idx & 511);
  int l = w >> 4, i = w & 15;
  int kt = t / ntiles, nt = t - kt * ntiles;
  int half = l >> 4;
  int klocal = (i < 8) ? (half * 8 + i) : (16 + half * 8 + (i - 8));
  int k = kt * 32 + klocal;
  int n = nt * 16 + (l & 15);
  float v = (k < Kreal) ? W[(size_t)k * N + n] : 0.0f;
  out[idx] = (_Float16)v;
}

// A fragment for v_wmma_f32_16x16x32_f16 from row-major LDS [16 x KPAD].
__device__ __forceinline__ v16h load_a_frag(const _Float16* __restrict__ A,
                                            int kpad, int kt, int lane) {
  const int m = lane & 15;
  const int half = lane >> 4;
  const _Float16* p = A + m * kpad + kt * 32 + half * 8;
  v8h lo = *(const v8h*)(p);
  v8h hi = *(const v8h*)(p + 16);
  v16h a;
#pragma unroll
  for (int i = 0; i < 8; ++i) { a[i] = lo[i]; a[i + 8] = hi[i]; }
  return a;
}

template <int KPAD>
__device__ __forceinline__ v8f wmma_tile(const _Float16* __restrict__ Arow,
                                         const _Float16* __restrict__ W,
                                         int nt, int ntiles, float binit, int lane) {
  v8f acc;
#pragma unroll
  for (int j = 0; j < 8; ++j) acc[j] = binit;
#pragma unroll
  for (int kt = 0; kt < KPAD / 32; ++kt) {
    v16h a = load_a_frag(Arow, KPAD, kt, lane);
    const _Float16* bp = W + (((size_t)(kt * ntiles + nt)) << 9) + (lane << 4);
    v16h b = *(const v16h*)bp;
    acc = __builtin_amdgcn_wmma_f32_16x16x32_f16(false, a, false, b,
                                                 (short)0, acc, false, false);
  }
  return acc;
}

// ---------------------------------------------------------------------------
// Encoder: gather 64 edge rows -> LDS, 2-layer MLP in WMMA, scatter-add to enc
// ---------------------------------------------------------------------------
__global__ __launch_bounds__(256) void k_encode(
    const float* __restrict__ obj_x, const float* __restrict__ obj_pos,
    const float* __restrict__ agent_pos, const int* __restrict__ eidx,
    const _Float16* __restrict__ W1, const float* __restrict__ b1,
    const _Float16* __restrict__ W2, const float* __restrict__ b2,
    float* __restrict__ enc, int E1) {
  __shared__ __align__(16) _Float16 sA[64 * K1PAD];
  __shared__ __align__(16) _Float16 sH[64 * EMB];
  __shared__ int sAgt[64];
  __shared__ int sObj[64];

  const int tid = threadIdx.x;
  const int lane = tid & 31;
  const int wave = tid >> 5;
  const int base = blockIdx.x * 64;

  __builtin_prefetch(W1, 0, 1);

  if (tid < 64) {
    int e = base + tid;
    sAgt[tid] = eidx[e];        // observing agent
    sObj[tid] = eidx[E1 + e];   // observed object
  }
  __syncthreads();

  for (int idx = tid; idx < 64 * K1PAD; idx += 256) {
    int r = idx / K1PAD;
    int k = idx - r * K1PAD;
    int o = sObj[r], a = sAgt[r];
    float v;
    if (k < 64)       v = obj_x[(size_t)o * 64 + k];
    else if (k < D0R) v = obj_pos[(size_t)o * 2 + (k - 64)] -
                          agent_pos[(size_t)a * 2 + (k - 64)];
    else              v = 0.0f;
    sA[idx] = (_Float16)v;
  }
  __syncthreads();

  // layer 1: relu(msg @ W1 + b1) -> sH (f16)
  for (int t = wave; t < 64; t += 8) {
    int mt = t >> 4, nt = t & 15;
    float bn = b1[nt * 16 + (lane & 15)];
    v8f acc = wmma_tile<K1PAD>(sA + mt * 16 * K1PAD, W1, nt, 16, bn, lane);
    int n = nt * 16 + (lane & 15);
    int mb = (lane >> 4) * 8;
#pragma unroll
    for (int j = 0; j < 8; ++j) {
      float v = acc[j];
      v = v > 0.0f ? v : 0.0f;
      sH[(mt * 16 + mb + j) * EMB + n] = (_Float16)v;
    }
  }
  __syncthreads();

  // layer 2: h @ W2 + b2, scatter-sum into enc[agent]
  for (int t = wave; t < 64; t += 8) {
    int mt = t >> 4, nt = t & 15;
    float bn = b2[nt * 16 + (lane & 15)];
    v8f acc = wmma_tile<EMB>(sH + mt * 16 * EMB, W2, nt, 16, bn, lane);
    int n = nt * 16 + (lane & 15);
    int mb = (lane >> 4) * 8;
#pragma unroll
    for (int j = 0; j < 8; ++j) {
      int aidx = sAgt[mt * 16 + mb + j];
      unsafeAtomicAdd(&enc[(size_t)aidx * EMB + n], acc[j]);
    }
  }
}

// ---------------------------------------------------------------------------
// Comm messages: async-gather [ench[src] || rel] -> LDS, MLP, scatter to agg
// ---------------------------------------------------------------------------
__global__ __launch_bounds__(256) void k_comm(
    const float* __restrict__ agent_pos, const int* __restrict__ eidx,
    const _Float16* __restrict__ ench,
    const _Float16* __restrict__ Wm, const float* __restrict__ bm,
    float* __restrict__ agg, int E2) {
  __shared__ __align__(16) _Float16 sA[64 * KMPAD];
  __shared__ int sSrc[64];
  __shared__ int sDst[64];

  const int tid = threadIdx.x;
  const int lane = tid & 31;
  const int wave = tid >> 5;
  const int base = blockIdx.x * 64;

  __builtin_prefetch(Wm, 0, 1);

  if (tid < 64) {
    int e = base + tid;
    sSrc[tid] = eidx[e];
    sDst[tid] = eidx[E2 + e];
  }
  __syncthreads();

#if defined(HAS_ASYNC_COPY)
  // per-row async DMA: one wave instruction moves one 512B ench row into LDS
  for (int i = 0; i < 8; ++i) {
    int r = wave + i * 8;
    int s = sSrc[r];
    const char* g = (const char*)(ench + (size_t)s * EMB) + lane * 16;
    char* l = (char*)sA + r * (KMPAD * 2) + lane * 16;
    __builtin_amdgcn_global_load_async_to_lds_b128(GPTR(g), LPTR(l), 0, 0);
  }
  // tail columns: rel (2) + zero padding (30)
  for (int idx = tid; idx < 64 * 32; idx += 256) {
    int r = idx >> 5;
    int k = EMB + (idx & 31);
    float v = 0.0f;
    if (k < EMB + 2) {
      int s = sSrc[r], d = sDst[r];
      v = agent_pos[(size_t)s * 2 + (k - EMB)] -
          agent_pos[(size_t)d * 2 + (k - EMB)];
    }
    sA[r * KMPAD + k] = (_Float16)v;
  }
  asm volatile("s_wait_asynccnt 0" ::: "memory");
#else
  for (int idx = tid; idx < 64 * KMPAD; idx += 256) {
    int r = idx / KMPAD;
    int k = idx - r * KMPAD;
    int s = sSrc[r], d = sDst[r];
    _Float16 v;
    if (k < EMB)          v = ench[(size_t)s * EMB + k];
    else if (k < EMB + 2) v = (_Float16)(agent_pos[(size_t)s * 2 + (k - EMB)] -
                                         agent_pos[(size_t)d * 2 + (k - EMB)]);
    else                  v = (_Float16)0.0f;
    sA[idx] = v;
  }
#endif
  __syncthreads();

  for (int t = wave; t < 64; t += 8) {
    int mt = t >> 4, nt = t & 15;
    float bn = bm[nt * 16 + (lane & 15)];
    v8f acc = wmma_tile<KMPAD>(sA + mt * 16 * KMPAD, Wm, nt, 16, bn, lane);
    int n = nt * 16 + (lane & 15);
    int mb = (lane >> 4) * 8;
#pragma unroll
    for (int j = 0; j < 8; ++j) {
      float v = acc[j];
      v = v > 0.0f ? v : 0.0f;
      int didx = sDst[mt * 16 + mb + j];
      unsafeAtomicAdd(&agg[(size_t)didx * EMB + n], v);
    }
  }
}

// ---------------------------------------------------------------------------
// Update: x = relu([ench || aggh] @ Wu + bu) -> xh (f16)
// ---------------------------------------------------------------------------
__global__ __launch_bounds__(256) void k_update(
    const _Float16* __restrict__ ench, const _Float16* __restrict__ aggh,
    const _Float16* __restrict__ Wu, const float* __restrict__ bu,
    _Float16* __restrict__ xh, int A) {
  __shared__ __align__(16) _Float16 sA[32 * KUPAD]; // 32 KB

  const int tid = threadIdx.x;
  const int lane = tid & 31;
  const int wave = tid >> 5;
  const int base = blockIdx.x * 32;

  __builtin_prefetch(Wu, 0, 1);

#if defined(HAS_ASYNC_COPY)
  for (int i = 0; i < 4; ++i) {
    int r = wave + i * 8;                       // 0..31
    const char* ge = (const char*)(ench + (size_t)(base + r) * EMB) + lane * 16;
    const char* ga = (const char*)(aggh + (size_t)(base + r) * EMB) + lane * 16;
    char* le = (char*)sA + r * 1024 + lane * 16;
    __builtin_amdgcn_global_load_async_to_lds_b128(GPTR(ge), LPTR(le), 0, 0);
    __builtin_amdgcn_global_load_async_to_lds_b128(GPTR(ga), LPTR(le + 512), 0, 0);
  }
  asm volatile("s_wait_asynccnt 0" ::: "memory");
#else
  for (int idx = tid; idx < 32 * KUPAD; idx += 256) {
    int r = idx >> 9;
    int k = idx & 511;
    int a = base + r;
    sA[idx] = (k < EMB) ? ench[(size_t)a * EMB + k]
                        : aggh[(size_t)a * EMB + (k - EMB)];
  }
#endif
  __syncthreads();

  for (int t = wave; t < 32; t += 8) {
    int mt = t >> 4, nt = t & 15;
    float bn = bu[nt * 16 + (lane & 15)];
    v8f acc = wmma_tile<KUPAD>(sA + mt * 16 * KUPAD, Wu, nt, 16, bn, lane);
    int n = nt * 16 + (lane & 15);
    int mb = (lane >> 4) * 8;
#pragma unroll
    for (int j = 0; j < 8; ++j) {
      float v = acc[j];
      v = v > 0.0f ? v : 0.0f;
      xh[(size_t)(base + mt * 16 + mb + j) * EMB + n] = (_Float16)v;
    }
  }
}

// ---------------------------------------------------------------------------
// Decoder: decoded = x @ Wd + bd -> d_out[A, 1056] (f32). A-tile staged by TDM.
// ---------------------------------------------------------------------------
__global__ __launch_bounds__(256) void k_decode(
    const _Float16* __restrict__ xh,
    const _Float16* __restrict__ Wd, const float* __restrict__ bd,
    float* __restrict__ out, int A) {
  __shared__ __align__(16) _Float16 sA[64 * EMB]; // 32 KB

  const int tid = threadIdx.x;
  const int lane = tid & 31;
  const int wave = tid >> 5;
  const int base = blockIdx.x * 64;

  __builtin_prefetch(Wd, 0, 1);

#if defined(HAS_TDM)
  if (wave == 0) {
    // D# group0: count=1 | lds_addr | 57-bit global addr | type=2
    unsigned lds_base =
        (unsigned)(uintptr_t)(__attribute__((address_space(3))) _Float16*)sA;
    unsigned long long ga =
        (unsigned long long)(uintptr_t)(xh + (size_t)base * EMB);
    u32x4 g0;
    g0[0] = 1u;
    g0[1] = lds_base;
    g0[2] = (unsigned)ga;
    g0[3] = (unsigned)((ga >> 32) & 0x01FFFFFFu) | 0x80000000u;
    // D# group1: data_size=2B; tensor 256 x A (f16 units); tile 256 x 64
    i32x8 g1;
    g1[0] = (int)(1u << 16);                       // data_size = 2 bytes
    g1[1] = (int)(256u << 16);                     // tensor_dim0 lo16 -> [31:16]
    g1[2] = (int)((((unsigned)A) & 0xffffu) << 16); // dim0 hi16 | dim1 lo16
    g1[3] = (int)((((unsigned)A) >> 16) | (256u << 16)); // dim1 hi | tile_dim0
    g1[4] = 64;                                    // tile_dim1 (rows)
    g1[5] = 256;                                   // tensor_dim0_stride lo32
    g1[6] = 0;
    g1[7] = 0;
    i32x4 g2 = {0, 0, 0, 0};
    i32x4 g3 = {0, 0, 0, 0};
#if defined(__clang_major__) && (__clang_major__ >= 23)
    i32x8 g4 = {0, 0, 0, 0, 0, 0, 0, 0};
    __builtin_amdgcn_tensor_load_to_lds(g0, g1, g2, g3, g4, 0);
#else
    __builtin_amdgcn_tensor_load_to_lds(g0, g1, g2, g3, 0);
#endif
    __builtin_amdgcn_s_wait_tensorcnt(0);
  }
#elif defined(HAS_ASYNC_COPY)
  for (int i = 0; i < 8; ++i) {
    int r = wave + i * 8;
    const char* g = (const char*)(xh + (size_t)(base + r) * EMB) + lane * 16;
    char* l = (char*)sA + r * 512 + lane * 16;
    __builtin_amdgcn_global_load_async_to_lds_b128(GPTR(g), LPTR(l), 0, 0);
  }
  asm volatile("s_wait_asynccnt 0" ::: "memory");
#else
  {
    const v8h* src = (const v8h*)(xh + (size_t)base * EMB);
    v8h* dstv = (v8h*)sA;
    for (int idx = tid; idx < 64 * EMB / 8; idx += 256) dstv[idx] = src[idx];
  }
#endif
  __syncthreads();

  for (int t = wave; t < 4 * NTDEC; t += 8) {
    int mt = t / NTDEC, nt = t - mt * NTDEC;
    float bn = bd[nt * 16 + (lane & 15)];
    v8f acc = wmma_tile<EMB>(sA + mt * 16 * EMB, Wd, nt, NTDEC, bn, lane);
    int n = nt * 16 + (lane & 15);
    int mb = (lane >> 4) * 8;
#pragma unroll
    for (int j = 0; j < 8; ++j) {
      out[(size_t)(base + mt * 16 + mb + j) * NDEC + n] = acc[j];
    }
  }
}

__global__ void k_batch(float* __restrict__ out, long long off, int total) {
  int i = blockIdx.x * blockDim.x + threadIdx.x;
  if (i < total) out[off + i] = (float)(i >> 4);
}

// ---------------------------------------------------------------------------
extern "C" void kernel_launch(void* const* d_in, const int* in_sizes, int n_in,
                              void* d_out, int out_size, void* d_ws, size_t ws_size,
                              hipStream_t stream) {
  const float* obj_x     = (const float*)d_in[0];
  const float* obj_pos   = (const float*)d_in[1];
  const float* agent_pos = (const float*)d_in[2];
  const int*   e_obs     = (const int*)d_in[3];
  const int*   e_comm    = (const int*)d_in[4];
  const float* W1  = (const float*)d_in[5];
  const float* b1  = (const float*)d_in[6];
  const float* W2  = (const float*)d_in[7];
  const float* b2  = (const float*)d_in[8];
  const float* Wm  = (const float*)d_in[9];
  const float* bm  = (const float*)d_in[10];
  const float* Wu  = (const float*)d_in[11];
  const float* bu  = (const float*)d_in[12];
  const float* Wd  = (const float*)d_in[13];
  const float* bd  = (const float*)d_in[14];
  // d_in[15]/d_in[16] (W_size/b_size) unused: reference returns (decoded, batch)

  const int E1 = in_sizes[3] / 2;
  const int E2 = in_sizes[4] / 2;
  const int A  = in_sizes[2] / 2;

  // workspace layout
  _Float16* w1h  = (_Float16*)d_ws;
  _Float16* w2h  = w1h + (size_t)K1PAD * EMB;
  _Float16* wmh  = w2h + (size_t)EMB * EMB;
  _Float16* wuh  = wmh + (size_t)KMPAD * EMB;
  _Float16* wdh  = wuh + (size_t)KUPAD * EMB;
  float*    enc  = (float*)(wdh + (size_t)EMB * NDEC);
  float*    agg  = enc + (size_t)A * EMB;
  _Float16* ench = (_Float16*)(agg + (size_t)A * EMB);
  _Float16* aggh = ench + (size_t)A * EMB;
  _Float16* xh   = aggh + (size_t)A * EMB;

  float* out = (float*)d_out;

  // 1) zero the scatter accumulators
  {
    long long n = 2LL * A * EMB;
    k_zero_f32<<<(int)((n + 255) / 256), 256, 0, stream>>>(enc, n);
  }

  // 2) convert + swizzle weights to f16 tiles
  {
    long long t1 = (long long)K1PAD * EMB;
    long long t2 = (long long)EMB * EMB;
    long long tm = (long long)KMPAD * EMB;
    long long tu = (long long)KUPAD * EMB;
    long long td = (long long)EMB * NDEC;
    k_convert<<<(int)((t1 + 255) / 256), 256, 0, stream>>>(W1, w1h, D0R,     EMB,  t1);
    k_convert<<<(int)((t2 + 255) / 256), 256, 0, stream>>>(W2, w2h, EMB,     EMB,  t2);
    k_convert<<<(int)((tm + 255) / 256), 256, 0, stream>>>(Wm, wmh, EMB + 2, EMB,  tm);
    k_convert<<<(int)((tu + 255) / 256), 256, 0, stream>>>(Wu, wuh, 2 * EMB, EMB,  tu);
    k_convert<<<(int)((td + 255) / 256), 256, 0, stream>>>(Wd, wdh, EMB,     NDEC, td);
  }

  // 3) encoder over observation edges (fused gather + MLP + scatter)
  k_encode<<<E1 / 64, 256, 0, stream>>>(obj_x, obj_pos, agent_pos, e_obs,
                                        w1h, b1, w2h, b2, enc, E1);

  // 4) enc -> f16 mirror for downstream gathers
  {
    long long n = (long long)A * EMB;
    k_f32_to_f16<<<(int)((n + 255) / 256), 256, 0, stream>>>(enc, ench, n);
  }

  // 5) message pass over comm edges (async gather + MLP + scatter)
  k_comm<<<E2 / 64, 256, 0, stream>>>(agent_pos, e_comm, ench, wmh, bm, agg, E2);

  // 6) agg -> f16 mirror
  {
    long long n = (long long)A * EMB;
    k_f32_to_f16<<<(int)((n + 255) / 256), 256, 0, stream>>>(agg, aggh, n);
  }

  // 7) per-agent update MLP
  k_update<<<A / 32, 256, 0, stream>>>(ench, aggh, wuh, bu, xh, A);

  // 8) set decoder -> d_out[:A*1056]
  k_decode<<<A / 64, 256, 0, stream>>>(xh, wdh, bd, out, A);

  // 9) batch ids -> d_out tail
  {
    int total = A * 16;
    long long off = (long long)A * NDEC;
    k_batch<<<(total + 255) / 256, 256, 0, stream>>>(out, off, total);
  }
}
